// QuantumAttentionLayer_74500502716747
// MI455X (gfx1250) — matrix operand order repaired and checked by reading, not compiled, not run on hardware
//
#include <hip/hip_runtime.h>
#include <hip/hip_bf16.h>

typedef __attribute__((ext_vector_type(16))) _Float16 v16h;
typedef __attribute__((ext_vector_type(8)))  float    v8f;
typedef unsigned __attribute__((ext_vector_type(4)))  u32x4;
typedef int      __attribute__((ext_vector_type(8)))  i32x8;

#define BB 4
#define SS 2048
#define EE 512
#define HH 8
#define DD 64
#define MM (BB*SS)      /* 8192 rows */
#define NQ 8
#define QDEPTH 3

// ---- WMMA helpers -----------------------------------------------------------

static __device__ __forceinline__ v8f wmma_f16(v16h a, v16h b, v8f c) {
  // emits v_wmma_f32_16x16x32_f16
  return __builtin_amdgcn_wmma_f32_16x16x32_f16(false, a, false, b, (short)0, c,
                                                false, false);
}

// A-matrix (16x32 f16) per-lane element -> K index (contiguous: k = 16*hi + e, matches B)
static __device__ __forceinline__ int a_k_of(int e, int hi) {
  return (hi ? 16 : 0) + e;
}
// B-matrix (32x16 f16): lanes 0-15 hold K=0..15 (e sequential), lanes 16-31 K=16..31
// -> per-lane B operand is 16 *contiguous* halves (32B) when the source is [n][k].

static __device__ __forceinline__ void wait_tensorcnt0() {
#if __has_builtin(__builtin_amdgcn_s_wait_tensorcnt)
  __builtin_amdgcn_s_wait_tensorcnt(0);
#else
  asm volatile("s_wait_tensorcnt 0x0" ::: "memory");
#endif
}

// ---- K1: Q/K/V projections (f32 in -> f16 head-major out) ------------------
// Block = 8 waves sharing one 16-column strip (nt): W tile staged in LDS once.

__global__ void qkv_proj_kernel(const float* __restrict__ q_in,
                                const float* __restrict__ k_in,
                                const float* __restrict__ v_in,
                                const float* __restrict__ Wq, const float* __restrict__ bq,
                                const float* __restrict__ Wk, const float* __restrict__ bk,
                                const float* __restrict__ Wv, const float* __restrict__ bv,
                                _Float16* __restrict__ Qd, _Float16* __restrict__ Kd,
                                _Float16* __restrict__ Vd) {
  __shared__ _Float16 Wt[2][16 * 32];     // [c][k] transposed, double-buffered
  const int tid = threadIdx.x;
  const int lane = tid & 31, wv = tid >> 5;
  const int hi = lane >> 4, col = lane & 15;
  const int mat = blockIdx.x / 2048;       // 0..2 -> Q/K/V
  const int rem = blockIdx.x % 2048;
  const int nt  = rem & 31;                // column tile, shared by block
  const int mt  = (rem >> 5) * 8 + wv;     // row tile per wave
  const float* X    = (mat == 0) ? q_in : (mat == 1) ? k_in : v_in;
  const float* W    = (mat == 0) ? Wq   : (mat == 1) ? Wk   : Wv;
  const float* bias = (mat == 0) ? bq   : (mat == 1) ? bk   : bv;
  _Float16* dst     = (mat == 0) ? Qd   : (mat == 1) ? Kd   : Vd;

  const int arow = mt * 16 + col;
  const int ncol = nt * 16 + col;

  auto stageW = [&](int buf, int k0) {     // 512 f16, 2 per thread
#pragma unroll
    for (int u = 0; u < 2; ++u) {
      int idx = tid * 2 + u;               // 0..511
      int kk = idx & 31, c = idx >> 5;
      Wt[buf][c * 32 + kk] = (_Float16)W[(size_t)(k0 + kk) * EE + nt * 16 + c];
    }
  };

  stageW(0, 0);
  v8f acc = {};
  for (int it = 0; it < EE / 32; ++it) {
    const int k0 = it * 32, cur = it & 1;
    __syncthreads();                        // Wt[cur] ready; prior reads done
    if (it + 1 < EE / 32) stageW(1 - cur, k0 + 32);
    v16h a;
#pragma unroll
    for (int e = 0; e < 16; ++e)
      a[e] = (_Float16)X[(size_t)arow * EE + k0 + a_k_of(e, hi)];
    if (k0 + 32 < EE) __builtin_prefetch(&X[(size_t)arow * EE + k0 + 32], 0, 1);
    v16h b = *(const v16h*)&Wt[cur][col * 32 + 16 * hi];
    acc = wmma_f16(a, b, acc);
  }
  const float bn = bias[ncol];
  const int h = ncol >> 6, d = ncol & 63;
#pragma unroll
  for (int r = 0; r < 8; ++r) {
    int m = mt * 16 + r + 8 * hi;
    int b_ = m / SS, s = m % SS;
    dst[(((size_t)b_ * HH + h) * SS + s) * DD + d] = (_Float16)(acc[r] + bn);
  }
}

// ---- K2: flash attention ----------------------------------------------------
// Block = 8 waves sharing (b,h); K panel staged via TDM (tensor_load_to_lds),
// V panel staged via global_load_async_to_lds_b128. Double-buffered.

#define NKB (SS / 32)   /* 64 key-block iterations */

__global__ void flash_attn_kernel(const _Float16* __restrict__ Qd,
                                  const _Float16* __restrict__ Kd,
                                  const _Float16* __restrict__ Vd,
                                  float* __restrict__ O,
                                  float* __restrict__ Mr, float* __restrict__ Lr) {
  __shared__ _Float16 Kt[2][32 * 64];      // [key][d], 4KB each
  __shared__ _Float16 Vt[2][32 * 64];      // [key][d], 4KB each
  __shared__ _Float16 Pst[8][16 * 32];     // per-wave P relayout
  const int tid = threadIdx.x;
  const int lane = tid & 31, wv = tid >> 5;
  const int hi = lane >> 4, col = lane & 15;
  const int bh = blockIdx.x >> 4;          // 0..31
  const int qt = (blockIdx.x & 15) * 8 + wv;
  const _Float16* Qb = Qd + (size_t)bh * SS * DD;
  const _Float16* Kb = Kd + (size_t)bh * SS * DD;
  const _Float16* Vb = Vd + (size_t)bh * SS * DD;

  auto stage_panels = [&](int buf, int kb) {
    if (wv == 0) {
      // Tensor DMA: 2D tile, 32 rows x 64 f16, row stride 64 elements
      unsigned ldsa = (unsigned)(size_t)&Kt[buf][0];   // flat->LDS: low bits
      unsigned long long ga = (unsigned long long)(size_t)(Kb + (size_t)kb * DD);
      u32x4 g0;
      g0[0] = 1u;                                      // count=1, user mode
      g0[1] = ldsa;                                    // lds_addr
      g0[2] = (unsigned)ga;                            // global_addr[31:0]
      g0[3] = ((unsigned)(ga >> 32) & 0x01ffffffu) | (2u << 30); // addr hi | type=2
      i32x8 g1;
      g1[0] = 1 << 16;                                 // data_size = 2 bytes
      g1[1] = DD << 16;                                // tensor_dim0 = 64 (lo16)
      g1[2] = (SS & 0xffff) << 16;                     // dim0 hi=0 | tensor_dim1 lo16
      g1[3] = DD << 16;                                // dim1 hi=0 | tile_dim0 = 64
      g1[4] = 32;                                      // tile_dim1 = 32, tile_dim2 = 0
      g1[5] = DD;                                      // tensor_dim0_stride = 64
      g1[6] = 0; g1[7] = 0;
      asm volatile("tensor_load_to_lds %0, %1" :: "s"(g0), "s"(g1) : "memory");
    }
    // V panel: 256 threads x 16B async copies
    int key = tid >> 3, c = tid & 7;
    unsigned ldsa = (unsigned)(size_t)&Vt[buf][key * DD + c * 8];
    unsigned long long ga =
        (unsigned long long)(size_t)(Vb + (size_t)(kb + key) * DD + c * 8);
    asm volatile("global_load_async_to_lds_b128 %0, %1, off"
                 :: "v"(ldsa), "v"(ga) : "memory");
  };

  // Q-tile A operands (loaded once)
  v16h qa0, qa1;
#pragma unroll
  for (int e = 0; e < 16; ++e) {
    int r = qt * 16 + col;
    qa0[e] = Qb[(size_t)r * DD + a_k_of(e, hi)];
    qa1[e] = Qb[(size_t)r * DD + 32 + a_k_of(e, hi)];
  }
  float mrow[8], lrow[8];
#pragma unroll
  for (int r = 0; r < 8; ++r) { mrow[r] = -1e30f; lrow[r] = 0.f; }
  v8f o0 = {}, o1 = {}, o2 = {}, o3 = {};

  stage_panels(0, 0);
  for (int i = 0; i < NKB; ++i) {
    const int kb0 = i * 32, cur = i & 1;
    asm volatile("s_wait_asynccnt 0x0" ::: "memory");
    if (wv == 0) wait_tensorcnt0();
    __syncthreads();                       // panels[cur] visible; prior reads done
    if (i + 1 < NKB) stage_panels(1 - cur, kb0 + 32);

    // ---- scores: 32 keys, two 16x16 tiles x 2 WMMAs over D=64 (from LDS)
    v8f s0 = {}, s1 = {};
    v16h kv;
    kv = *(const v16h*)&Kt[cur][col * 64 + 16 * hi];
    s0 = wmma_f16(qa0, kv, s0);
    kv = *(const v16h*)&Kt[cur][col * 64 + 32 + 16 * hi];
    s0 = wmma_f16(qa1, kv, s0);
    kv = *(const v16h*)&Kt[cur][(16 + col) * 64 + 16 * hi];
    s1 = wmma_f16(qa0, kv, s1);
    kv = *(const v16h*)&Kt[cur][(16 + col) * 64 + 32 + 16 * hi];
    s1 = wmma_f16(qa1, kv, s1);

    // ---- online softmax per row
#pragma unroll
    for (int r = 0; r < 8; ++r) {
      float t0 = s0[r] * 0.125f, t1 = s1[r] * 0.125f;  // 1/sqrt(64)
      float v = fmaxf(t0, t1);
      v = fmaxf(v, __shfl_xor(v, 1, 32));
      v = fmaxf(v, __shfl_xor(v, 2, 32));
      v = fmaxf(v, __shfl_xor(v, 4, 32));
      v = fmaxf(v, __shfl_xor(v, 8, 32));
      float mnew = fmaxf(mrow[r], v);
      float alpha = __expf(mrow[r] - mnew);
      lrow[r] *= alpha;
      o0[r] *= alpha; o1[r] *= alpha; o2[r] *= alpha; o3[r] *= alpha;
      mrow[r] = mnew;
      float p0 = __expf(t0 - mnew), p1 = __expf(t1 - mnew);
      float rs = p0 + p1;
      rs += __shfl_xor(rs, 1, 32);
      rs += __shfl_xor(rs, 2, 32);
      rs += __shfl_xor(rs, 4, 32);
      rs += __shfl_xor(rs, 8, 32);
      lrow[r] += rs;
      Pst[wv][(r + 8 * hi) * 32 + col]      = (_Float16)p0;
      Pst[wv][(r + 8 * hi) * 32 + 16 + col] = (_Float16)p1;
    }
    asm volatile("s_wait_dscnt 0x0" ::: "memory");   // wave-local LDS relayout
    v16h pa;
#pragma unroll
    for (int e = 0; e < 16; ++e) pa[e] = Pst[wv][col * 32 + a_k_of(e, hi)];

    // ---- P(16x32) x V(32x64) from LDS: 4 WMMAs
#pragma unroll
    for (int j = 0; j < 4; ++j) {
      v16h vb;
#pragma unroll
      for (int e = 0; e < 16; ++e)
        vb[e] = Vt[cur][(e + 16 * hi) * DD + j * 16 + col];
      if (j == 0)      o0 = wmma_f16(pa, vb, o0);
      else if (j == 1) o1 = wmma_f16(pa, vb, o1);
      else if (j == 2) o2 = wmma_f16(pa, vb, o2);
      else             o3 = wmma_f16(pa, vb, o3);
    }
  }

#pragma unroll
  for (int r = 0; r < 8; ++r) {
    int s = qt * 16 + r + 8 * hi;
    float linv = 1.0f / lrow[r];
    size_t base = ((size_t)bh * SS + s) * DD;
    O[base + col]      = o0[r] * linv;
    O[base + 16 + col] = o1[r] * linv;
    O[base + 32 + col] = o2[r] * linv;
    O[base + 48 + col] = o3[r] * linv;
    if (col == 0) { Mr[(size_t)bh * SS + s] = mrow[r]; Lr[(size_t)bh * SS + s] = lrow[r]; }
  }
}

// ---- K3: diagonal attention probability ------------------------------------

__global__ void diag_pat_kernel(const _Float16* __restrict__ Qd,
                                const _Float16* __restrict__ Kd,
                                const float* __restrict__ Mr, const float* __restrict__ Lr,
                                float* __restrict__ pat) {
  int idx = blockIdx.x * 256 + threadIdx.x;     // over BB*HH*SS
  int bh = idx / SS, s = idx % SS;
  int b = bh / HH, h = bh % HH;
  const _Float16* q = Qd + ((size_t)bh * SS + s) * DD;
  const _Float16* k = Kd + ((size_t)bh * SS + s) * DD;
  float dot = 0.f;
#pragma unroll 16
  for (int d = 0; d < DD; ++d) dot += (float)q[d] * (float)k[d];
  dot *= 0.125f;
  float p = __expf(dot - Mr[(size_t)bh * SS + s]) / Lr[(size_t)bh * SS + s];
  pat[((size_t)b * SS + s) * HH + h] = p;
}

// ---- K4: 8-qubit state-vector sim, one wave per token ----------------------

struct C2 { float r, i; };
static __device__ __forceinline__ C2 cmul(C2 a, C2 b) {
  return C2{a.r * b.r - a.i * b.i, a.r * b.i + a.i * b.r};
}
static __device__ __forceinline__ C2 cadd(C2 a, C2 b) { return C2{a.r + b.r, a.i + b.i}; }

__global__ void quantum_kernel(const float* __restrict__ pat,
                               const float* __restrict__ encW, const float* __restrict__ encB,
                               const float* __restrict__ big,  const float* __restrict__ bib,
                               const float* __restrict__ rot,  const float* __restrict__ ent,
                               const float* __restrict__ decW, const float* __restrict__ decB,
                               const float* __restrict__ bog,  const float* __restrict__ bob,
                               float* __restrict__ enh) {
  const int lane = threadIdx.x & 31, wv = threadIdx.x >> 5;
  const int n = blockIdx.x * 8 + wv;           // token/state index

  float pv[NQ], ev[NQ];
#pragma unroll
  for (int i = 0; i < NQ; ++i) pv[i] = pat[(size_t)n * NQ + i];
#pragma unroll
  for (int j = 0; j < NQ; ++j) {
    float acc = encB[j];
#pragma unroll
    for (int i = 0; i < NQ; ++i) acc += pv[i] * encW[i * NQ + j];
    ev[j] = tanhf(big[j] * acc + bib[j]);
  }
  float nrm2 = 0.f;
#pragma unroll
  for (int j = 0; j < NQ; ++j) nrm2 += ev[j] * ev[j];
  float nrm = sqrtf(nrm2);
  if (nrm > 0.f) {
    float inv = 1.0f / fmaxf(nrm, 1e-30f);
#pragma unroll
    for (int j = 0; j < NQ; ++j) ev[j] *= inv;
  }
  float v0[NQ], v1[NQ];
#pragma unroll
  for (int j = 0; j < NQ; ++j) {
    float c = fminf(fabsf(ev[j]), 1.0f);
    v0[j] = sqrtf(fmaxf(1.0f - c * c, 0.0f));
    v1[j] = copysignf(c, ev[j]);
  }
  C2 amp[8];
#pragma unroll
  for (int j = 0; j < 8; ++j) {
    int aidx = lane * 8 + j;
    float pr = 1.0f;
#pragma unroll
    for (int i = 0; i < NQ; ++i) pr *= ((aidx >> (7 - i)) & 1) ? v1[i] : v0[i];
    amp[j] = C2{pr, 0.0f};
  }

  for (int l = 0; l < QDEPTH; ++l) {
#pragma unroll
    for (int q = 0; q < NQ; ++q) {               // U = Rz*Ry*Rx on qubit q
      const float* rp = rot + (l * NQ + q) * 3;
      float tx = 0.5f * rp[0], ty = 0.5f * rp[1], pz = 0.5f * rp[2];
      float cx = __cosf(tx), sx = __sinf(tx);
      float cy = __cosf(ty), sy = __sinf(ty);
      float cz = __cosf(pz), sz = __sinf(pz);
      C2 m00{cy * cx,  sy * sx}, m01{-sy * cx, -cy * sx};
      C2 m10{sy * cx, -cy * sx}, m11{ cy * cx, -sy * sx};
      C2 e0{cz, -sz}, e1{cz, sz};
      C2 u00 = cmul(e0, m00), u01 = cmul(e0, m01);
      C2 u10 = cmul(e1, m10), u11 = cmul(e1, m11);
      const int bp = 7 - q;
      if (bp < 3) {
        const int tm = 1 << bp;
#pragma unroll
        for (int j = 0; j < 8; ++j) if ((j & tm) == 0) {
          C2 x0 = amp[j], x1 = amp[j | tm];
          amp[j]      = cadd(cmul(u00, x0), cmul(u01, x1));
          amp[j | tm] = cadd(cmul(u10, x0), cmul(u11, x1));
        }
      } else {
        const int lm = 1 << (bp - 3);
        const bool hib = (lane & lm) != 0;
#pragma unroll
        for (int j = 0; j < 8; ++j) {
          C2 oth; oth.r = __shfl_xor(amp[j].r, lm, 32);
          oth.i = __shfl_xor(amp[j].i, lm, 32);
          C2 mine = amp[j];
          amp[j] = hib ? cadd(cmul(u10, oth), cmul(u11, mine))
                       : cadd(cmul(u00, mine), cmul(u01, oth));
        }
      }
    }
#pragma unroll
    for (int i = 0; i < NQ - 1; ++i) {            // CRy(control=i, target=i+1)
      float tc = 0.5f * ent[l * (NQ - 1) + i];
      float cc = __cosf(tc), sn = __sinf(tc);
      const int cb = 7 - i, tb = 6 - i;
      if (tb >= 3) {
        const int lm = 1 << (tb - 3);
        const bool ctrl = (lane & (1 << (cb - 3))) != 0;
        const bool hib = (lane & lm) != 0;
#pragma unroll
        for (int j = 0; j < 8; ++j) {
          float orr = __shfl_xor(amp[j].r, lm, 32);
          float oii = __shfl_xor(amp[j].i, lm, 32);
          if (ctrl) {
            if (hib) { amp[j].r = sn * orr + cc * amp[j].r; amp[j].i = sn * oii + cc * amp[j].i; }
            else     { amp[j].r = cc * amp[j].r - sn * orr; amp[j].i = cc * amp[j].i - sn * oii; }
          }
        }
      } else if (cb >= 3) {                       // i==4
        const bool ctrl = (lane & (1 << (cb - 3))) != 0;
        const int tm = 1 << tb;
#pragma unroll
        for (int j = 0; j < 8; ++j) if ((j & tm) == 0) {
          C2 x0 = amp[j], x1 = amp[j | tm];
          if (ctrl) {
            amp[j].r = cc * x0.r - sn * x1.r;      amp[j].i = cc * x0.i - sn * x1.i;
            amp[j | tm].r = sn * x0.r + cc * x1.r; amp[j | tm].i = sn * x0.i + cc * x1.i;
          }
        }
      } else {                                    // i==5,6
        const int cm = 1 << cb, tm = 1 << tb;
#pragma unroll
        for (int j = 0; j < 8; ++j) if ((j & cm) && !(j & tm)) {
          C2 x0 = amp[j], x1 = amp[j | tm];
          amp[j].r = cc * x0.r - sn * x1.r;      amp[j].i = cc * x0.i - sn * x1.i;
          amp[j | tm].r = sn * x0.r + cc * x1.r; amp[j | tm].i = sn * x0.i + cc * x1.i;
        }
      }
    }
  }

  float p[8], psum = 0.f;
#pragma unroll
  for (int j = 0; j < 8; ++j) { p[j] = amp[j].r * amp[j].r + amp[j].i * amp[j].i; psum += p[j]; }
  float meas[NQ];
#pragma unroll
  for (int i = 0; i < NQ; ++i) {
    const int bp = 7 - i;
    float part;
    if (bp < 3) {
      const int tm = 1 << bp; part = 0.f;
#pragma unroll
      for (int j = 0; j < 8; ++j) if (j & tm) part += p[j];
    } else {
      part = (lane & (1 << (bp - 3))) ? psum : 0.0f;
    }
    part += __shfl_xor(part, 16, 32);
    part += __shfl_xor(part, 8, 32);
    part += __shfl_xor(part, 4, 32);
    part += __shfl_xor(part, 2, 32);
    part += __shfl_xor(part, 1, 32);
    meas[i] = part;
  }
  if (lane < HH) {
    const int h = lane;
    float acc = decB[h];
#pragma unroll
    for (int i = 0; i < NQ; ++i) acc += meas[i] * decW[i * HH + h];
    enh[(size_t)n * HH + h] = bog[h] * acc + bob[h];
  }
}

// ---- K5: enh-scale + output projection + residual (Wo staged in LDS) -------

__global__ void outproj_kernel(const float* __restrict__ O, const float* __restrict__ enh,
                               const float* __restrict__ Wo, const float* __restrict__ bo,
                               const float* __restrict__ query, float* __restrict__ tmp) {
  __shared__ _Float16 Wt[2][16 * 32];
  const int tid = threadIdx.x;
  const int lane = tid & 31, wv = tid >> 5;
  const int hi = lane >> 4, col = lane & 15;
  const int nt = blockIdx.x & 31;
  const int mt = (blockIdx.x >> 5) * 8 + wv;
  const int arow = mt * 16 + col;
  const int b_ = arow / SS, s = arow % SS;
  const int ncol = nt * 16 + col;

  auto stageW = [&](int buf, int k0) {
#pragma unroll
    for (int u = 0; u < 2; ++u) {
      int idx = tid * 2 + u;
      int kk = idx & 31, c = idx >> 5;
      Wt[buf][c * 32 + kk] = (_Float16)Wo[(size_t)(k0 + kk) * EE + nt * 16 + c];
    }
  };

  stageW(0, 0);
  v8f acc = {};
  for (int it = 0; it < EE / 32; ++it) {
    const int k0 = it * 32, cur = it & 1;
    __syncthreads();
    if (it + 1 < EE / 32) stageW(1 - cur, k0 + 32);
    v16h a;
#pragma unroll
    for (int e = 0; e < 16; ++e) {
      int k = k0 + a_k_of(e, hi);
      int h = k >> 6, d = k & 63;
      float x = O[(((size_t)b_ * HH + h) * SS + s) * DD + d] * enh[(size_t)arow * HH + h];
      a[e] = (_Float16)x;
    }
    v16h b = *(const v16h*)&Wt[cur][col * 32 + 16 * hi];
    acc = wmma_f16(a, b, acc);
  }
  const float bn = bo[ncol];
#pragma unroll
  for (int r = 0; r < 8; ++r) {
    int m = mt * 16 + r + 8 * hi;
    tmp[(size_t)m * EE + ncol] = acc[r] + bn + query[(size_t)m * EE + ncol];
  }
}

// ---- K6: LayerNorm, one wave per row ---------------------------------------

__global__ void layernorm_kernel(const float* __restrict__ tmp, const float* __restrict__ g,
                                 const float* __restrict__ bta, float* __restrict__ out) {
  const int lane = threadIdx.x & 31, wv = threadIdx.x >> 5;
  const int m = blockIdx.x * 8 + wv;
  float x[16], sum = 0.f;
#pragma unroll
  for (int t = 0; t < 16; ++t) { x[t] = tmp[(size_t)m * EE + t * 32 + lane]; sum += x[t]; }
  sum += __shfl_xor(sum, 16, 32);
  sum += __shfl_xor(sum, 8, 32);
  sum += __shfl_xor(sum, 4, 32);
  sum += __shfl_xor(sum, 2, 32);
  sum += __shfl_xor(sum, 1, 32);
  const float mu = sum * (1.0f / EE);
  float s2 = 0.f;
#pragma unroll
  for (int t = 0; t < 16; ++t) { float d = x[t] - mu; s2 += d * d; }
  s2 += __shfl_xor(s2, 16, 32);
  s2 += __shfl_xor(s2, 8, 32);
  s2 += __shfl_xor(s2, 4, 32);
  s2 += __shfl_xor(s2, 2, 32);
  s2 += __shfl_xor(s2, 1, 32);
  const float rstd = rsqrtf(s2 * (1.0f / EE) + 1e-5f);
#pragma unroll
  for (int t = 0; t < 16; ++t) {
    int e = t * 32 + lane;
    out[(size_t)m * EE + e] = g[e] * (x[t] - mu) * rstd + bta[e];
  }
}

// ---- host-side launch ------------------------------------------------------

extern "C" void kernel_launch(void* const* d_in, const int* in_sizes, int n_in,
                              void* d_out, int out_size, void* d_ws, size_t ws_size,
                              hipStream_t stream) {
  (void)in_sizes; (void)n_in; (void)out_size; (void)ws_size;
  const float* query = (const float*)d_in[0];
  const float* key_t = (const float*)d_in[1];
  const float* value = (const float*)d_in[2];
  const float* Wq = (const float*)d_in[3];  const float* bq = (const float*)d_in[4];
  const float* Wk = (const float*)d_in[5];  const float* bk = (const float*)d_in[6];
  const float* Wv = (const float*)d_in[7];  const float* bv = (const float*)d_in[8];
  const float* Wo = (const float*)d_in[9];  const float* bo = (const float*)d_in[10];
  const float* encW = (const float*)d_in[11]; const float* encB = (const float*)d_in[12];
  const float* big  = (const float*)d_in[13]; const float* bib  = (const float*)d_in[14];
  const float* rot  = (const float*)d_in[15]; const float* ent  = (const float*)d_in[16];
  const float* decW = (const float*)d_in[17]; const float* decB = (const float*)d_in[18];
  const float* bog  = (const float*)d_in[19]; const float* bob  = (const float*)d_in[20];
  const float* lng  = (const float*)d_in[21]; const float* lnb  = (const float*)d_in[22];
  float* out = (float*)d_out;

  char* ws = (char*)d_ws;
  size_t off = 0;
  auto alloc = [&](size_t bytes) -> void* {
    void* p = ws + off;
    off += (bytes + 255) & ~(size_t)255;
    return p;
  };
  const size_t nQKV = (size_t)MM * EE;          // == B*H*S*D
  _Float16* Qd = (_Float16*)alloc(nQKV * 2);
  _Float16* Kd = (_Float16*)alloc(nQKV * 2);
  _Float16* Vd = (_Float16*)alloc(nQKV * 2);
  float* O    = (float*)alloc(nQKV * 4);
  float* Mr   = (float*)alloc((size_t)BB * HH * SS * 4);
  float* Lr   = (float*)alloc((size_t)BB * HH * SS * 4);
  float* pat  = (float*)alloc((size_t)MM * HH * 4);
  float* enh  = (float*)alloc((size_t)MM * HH * 4);
  float* tmp  = (float*)alloc((size_t)MM * EE * 4);

  qkv_proj_kernel<<<6144, 256, 0, stream>>>(query, key_t, value, Wq, bq, Wk, bk, Wv, bv,
                                            Qd, Kd, Vd);
  flash_attn_kernel<<<512, 256, 0, stream>>>(Qd, Kd, Vd, O, Mr, Lr);
  diag_pat_kernel<<<256, 256, 0, stream>>>(Qd, Kd, Mr, Lr, pat);
  quantum_kernel<<<1024, 256, 0, stream>>>(pat, encW, encB, big, bib, rot, ent,
                                           decW, decB, bog, bob, enh);
  outproj_kernel<<<2048, 256, 0, stream>>>(O, enh, Wo, bo, query, tmp);
  layernorm_kernel<<<1024, 256, 0, stream>>>(tmp, lng, lnb, out);
}
